// kopt_Decoder_63385127354442
// MI455X (gfx1250) — compile-verified, matrix operand order checked by reading, not run on hardware
//
#include <hip/hip_runtime.h>
#include <math.h>

#define BS   64
#define GSZ  2000
#define DD   128
#define KMAX 8
#define CLIPV 10.0f

typedef __attribute__((ext_vector_type(16))) __bf16 v16bf;
typedef __attribute__((ext_vector_type(8)))  float  v8f;
typedef __attribute__((ext_vector_type(8)))  __bf16 bf16x8;

__device__ __forceinline__ float fast_tanh(float x) {
#if __has_builtin(__builtin_amdgcn_tanhf)
  return __builtin_amdgcn_tanhf(x);
#else
  return tanhf(x);
#endif
}
__device__ __forceinline__ float fast_sigmoid(float x) {
  return 1.0f / (1.0f + __expf(-x));
}
__device__ __forceinline__ float dot128(const float* __restrict__ v,
                                        const float* __restrict__ w) {
  float s = 0.f;
#pragma unroll 8
  for (int e = 0; e < 128; ++e) s = fmaf(v[e], w[e], s);
  return s;
}

// ---------------------------------------------------------------------------
// Kernel A: K{1..4}h[b,g,:] = h[b,g,:] @ WK{1..4}^T, output bf16.
// M = BS*GSZ = 128000 rows, K = N = 128. One wave -> 16 rows x 128 cols.
// Uses v_wmma_f32_16x16x32_bf16. blockIdx.y selects which weight matrix.
// ---------------------------------------------------------------------------
__global__ __launch_bounds__(128) void kprep_wmma(
    const float* __restrict__ h,
    const float* __restrict__ W0, const float* __restrict__ W1,
    const float* __restrict__ W2, const float* __restrict__ W3,
    __bf16* __restrict__ O0, __bf16* __restrict__ O1,
    __bf16* __restrict__ O2, __bf16* __restrict__ O3) {
  const float* W;
  __bf16* O;
  switch (blockIdx.y) {
    case 0: W = W0; O = O0; break;
    case 1: W = W1; O = O1; break;
    case 2: W = W2; O = O2; break;
    default: W = W3; O = O3; break;
  }
  const int lane = threadIdx.x & 31;
  const int wave = threadIdx.x >> 5;
  const int m0   = (blockIdx.x * 4 + wave) * 16;
  const int mrow = m0 + (lane & 15);
  const int half = lane >> 4;

  v8f acc[8] = {};

  const float* hrow = h + (size_t)mrow * DD;
#pragma unroll
  for (int kc = 0; kc < 128; kc += 32) {
    // A fragment: 16x32 bf16, lane holds row (lane&15), two contiguous
    // 8-element K segments at kc+half*8 and kc+16+half*8 (ISA 16-bit A layout).
    v16bf afrag;
#pragma unroll
    for (int s = 0; s < 2; ++s) {
      const int kb = kc + s * 16 + half * 8;
#pragma unroll
      for (int j = 0; j < 8; ++j) afrag[s * 8 + j] = (__bf16)hrow[kb + j];
    }
#pragma unroll
    for (int nt = 0; nt < 8; ++nt) {
      const int n = nt * 16 + (lane & 15);
      const float* wrow = W + (size_t)n * DD;  // B col n == W row n (W is N x K)
      v16bf bfrag;
#pragma unroll
      for (int s = 0; s < 2; ++s) {
        const int kb = kc + s * 16 + half * 8;
#pragma unroll
        for (int j = 0; j < 8; ++j) bfrag[s * 8 + j] = (__bf16)wrow[kb + j];
      }
      acc[nt] = __builtin_amdgcn_wmma_f32_16x16x32_bf16(
          false, afrag, false, bfrag, (short)0, acc[nt], false, false);
    }
  }
  // C/D layout: VGPR r holds row half*8+r, lane&15 is the column.
#pragma unroll
  for (int nt = 0; nt < 8; ++nt) {
    const int n = nt * 16 + (lane & 15);
#pragma unroll
    for (int r = 0; r < 8; ++r) {
      const int m = m0 + half * 8 + r;
      O[(size_t)m * DD + n] = (__bf16)acc[nt][r];
    }
  }
}

// ---------------------------------------------------------------------------
// Kernel B: sequential decode. One block per batch row; 256 threads.
// All 8 steps run inside the block; state in LDS.
// ---------------------------------------------------------------------------
__global__ __launch_bounds__(256) void kdecode(
    const float* __restrict__ h, const int* __restrict__ rec,
    const int* __restrict__ visited_time, const int* __restrict__ last_action,
    const int* __restrict__ fixed_action,
    const float* __restrict__ WQ1, const float* __restrict__ WQ2,
    const float* __restrict__ WQ3, const float* __restrict__ WQ4,
    const float* __restrict__ V1, const float* __restrict__ V2,
    const float* __restrict__ ih_w, const float* __restrict__ ih_b,
    const float* __restrict__ init_query,
    const float* __restrict__ r1_wih, const float* __restrict__ r1_whh,
    const float* __restrict__ r1_bih, const float* __restrict__ r1_bhh,
    const float* __restrict__ r2_wih, const float* __restrict__ r2_whh,
    const float* __restrict__ r2_bih, const float* __restrict__ r2_bhh,
    const __bf16* __restrict__ K1h, const __bf16* __restrict__ K2h,
    const __bf16* __restrict__ K3h, const __bf16* __restrict__ K4h,
    int* __restrict__ out_actions, float* __restrict__ out_ll) {
  const int b = blockIdx.x;
  const int t = threadIdx.x;

  __shared__ float s_q1[DD], s_q2[DD], s_iq1[DD], s_iq2[DD];
  __shared__ float s_Q1[DD], s_Q2[DD], s_Q3[DD], s_Q4[DD];
  __shared__ float s_V1[DD], s_V2[DD];
  __shared__ float s_gi1[3 * DD], s_gh1[3 * DD], s_gi2[3 * DD], s_gh2[3 * DD];
  __shared__ float s_logits[2048];
  __shared__ float s_red[256], s_part[256];
  __shared__ int st_ai[KMAX], st_kal[KMAX + 1], st_kar[KMAX];
  __shared__ int s_stopped, s_nol, s_vt0, s_vthr, s_useg2, s_ov1, s_ov2;
  __shared__ int s_iq1src, s_iq2src;
  __shared__ float s_ll;

  // ---- prologue: hmean, q1=q2=hmean@ih_w.T+ih_b, iq=init_query ----
  {
    const int e = t & 127, gpar = t >> 7;
    float acc = 0.f;
    for (int g = gpar; g < GSZ; g += 2)
      acc += h[((size_t)b * GSZ + g) * DD + e];
    s_part[t] = acc;
  }
  if (t == 0) {
    s_stopped = 1; s_nol = -1; s_ll = 0.f;
    s_vt0 = 0; s_vthr = 0; s_useg2 = 0; s_ov1 = -1; s_ov2 = -1;
    for (int j = 0; j <= KMAX; ++j) st_kal[j] = 0;
    for (int j = 0; j < KMAX; ++j) { st_kar[j] = 0; st_ai[j] = 0; }
  }
  __syncthreads();
  if (t < DD) s_gi1[t] = (s_part[t] + s_part[t + 128]) * (1.0f / (float)GSZ);
  __syncthreads();
  if (t < DD) {
    const float q = dot128(s_gi1, ih_w + (size_t)t * DD) + ih_b[t];
    s_q1[t] = q; s_q2[t] = q;
    const float iq = init_query[t];
    s_iq1[t] = iq; s_iq2[t] = iq;
    s_V1[t] = V1[t]; s_V2[t] = V2[t];
  }
  __syncthreads();

  const int lastA = last_action[b];

  for (int i = 0; i < KMAX; ++i) {
    // ---- GRU gemvs: gi/gh for both cells (1536 dots of length 128) ----
    for (int idx = t; idx < 4 * 3 * DD; idx += 256) {
      const int which = idx / (3 * DD), o = idx % (3 * DD);
      float r;
      switch (which) {
        case 0: r = dot128(s_iq1, r1_wih + (size_t)o * DD) + r1_bih[o]; s_gi1[o] = r; break;
        case 1: r = dot128(s_q1,  r1_whh + (size_t)o * DD) + r1_bhh[o]; s_gh1[o] = r; break;
        case 2: r = dot128(s_iq2, r2_wih + (size_t)o * DD) + r2_bih[o]; s_gi2[o] = r; break;
        default:r = dot128(s_q2,  r2_whh + (size_t)o * DD) + r2_bhh[o]; s_gh2[o] = r; break;
      }
    }
    __syncthreads();
    // ---- GRU combine ----
    if (t < DD) {
      const int d = t;
      const float r = fast_sigmoid(s_gi1[d] + s_gh1[d]);
      const float z = fast_sigmoid(s_gi1[DD + d] + s_gh1[DD + d]);
      const float n = fast_tanh(s_gi1[2 * DD + d] + r * s_gh1[2 * DD + d]);
      s_q1[d] = (1.f - z) * n + z * s_q1[d];
    } else {
      const int d = t - DD;
      const float r = fast_sigmoid(s_gi2[d] + s_gh2[d]);
      const float z = fast_sigmoid(s_gi2[DD + d] + s_gh2[DD + d]);
      const float n = fast_tanh(s_gi2[2 * DD + d] + r * s_gh2[2 * DD + d]);
      s_q2[d] = (1.f - z) * n + z * s_q2[d];
    }
    __syncthreads();
    // ---- Q projections ----
    if (t < DD) {
      s_Q1[t] = dot128(s_q1, WQ1 + (size_t)t * DD);
      s_Q3[t] = dot128(s_q1, WQ3 + (size_t)t * DD);
    } else {
      const int d = t - DD;
      s_Q2[d] = dot128(s_q2, WQ2 + (size_t)d * DD);
      s_Q4[d] = dot128(s_q2, WQ4 + (size_t)d * DD);
    }
    __syncthreads();

    // ---- attention scan over GSZ (dominant memory traffic, bf16 keys) ----
    const int vthr = s_vthr, useg2 = s_useg2, ov1 = s_ov1, ov2 = s_ov2;
    const int vt0 = s_vt0;
    float lmax = -INFINITY;
    for (int g = t; g < GSZ; g += 256) {
      bool masked;
      if (i == 0) {
        masked = (g == lastA);
      } else {
        int vtg = visited_time[(size_t)b * GSZ + g] - vt0;
        if (vtg < 0) vtg += GSZ;
        masked = (vtg <= vthr) || (useg2 && vtg > GSZ - 2);
        if (g == ov1 || g == ov2) masked = false;
      }
      float lg;
      if (masked) {
        lg = -1e30f;
      } else {
        const size_t ro = ((size_t)b * GSZ + g) * DD;
        const bf16x8* p1 = (const bf16x8*)(K1h + ro);
        const bf16x8* p2 = (const bf16x8*)(K2h + ro);
        const bf16x8* p3 = (const bf16x8*)(K3h + ro);
        const bf16x8* p4 = (const bf16x8*)(K4h + ro);
        float s1 = 0.f, s2 = 0.f;
#pragma unroll 4
        for (int c = 0; c < 16; ++c) {
          const bf16x8 k1 = p1[c], k2 = p2[c], k3 = p3[c], k4 = p4[c];
#pragma unroll
          for (int j = 0; j < 8; ++j) {
            const int e = c * 8 + j;
            const float a1 = (float)k1[j] + s_Q1[e] + (float)k3[j] * s_Q3[e];
            s1 += s_V1[e] * fast_tanh(a1);
            const float a2 = (float)k2[j] + s_Q2[e] + (float)k4[j] * s_Q4[e];
            s2 += s_V2[e] * fast_tanh(a2);
          }
        }
        lg = fast_tanh(s1 + s2) * CLIPV;
      }
      s_logits[g] = lg;
      lmax = fmaxf(lmax, lg);
    }
    // ---- block max ----
    s_red[t] = lmax;
    __syncthreads();
    for (int s = 128; s > 0; s >>= 1) {
      if (t < s) s_red[t] = fmaxf(s_red[t], s_red[t + s]);
      __syncthreads();
    }
    const float maxv = s_red[0];
    // ---- block sum of exp ----
    float lsum = 0.f;
    for (int g = t; g < GSZ; g += 256) lsum += __expf(s_logits[g] - maxv);
    s_part[t] = lsum;
    __syncthreads();
    for (int s = 128; s > 0; s >>= 1) {
      if (t < s) s_part[t] += s_part[t + s];
      __syncthreads();
    }

    // ---- serial state update (thread 0) ----
    if (t == 0) {
      const int stopped_old = s_stopped;
      const int fa = fixed_action[b * KMAX + i];
      const int act = (i == 0) ? fa : (stopped_old ? st_ai[0] : fa);
      const float loss_now = s_logits[act] - maxv - __logf(s_part[0]);
      if (i == 0 || !stopped_old) s_ll += loss_now;
      const int nna = rec[(size_t)b * GSZ + act];
      st_ai[i] = act;
      const int im1 = (i == 0) ? (KMAX - 1) : (i - 1);
      if (stopped_old) st_kal[i] = act;
      if (!stopped_old) st_kar[im1] = act;
      st_kal[i + 1] = nna;
      // input sources use OLD stopped / OLD nol
      s_iq1src = act;
      s_iq2src = stopped_old ? act : (((s_nol % GSZ) + GSZ) % GSZ);
      int stopped = (i == 0) ? (act == s_nol) : (stopped_old || (act == s_nol));
      s_stopped = stopped;
      if (stopped) { st_kal[i] = st_kal[im1]; st_kar[i] = st_kar[im1]; }
      if (i == 0) s_vt0 = visited_time[(size_t)b * GSZ + act];
      int va = visited_time[(size_t)b * GSZ + act] - s_vt0;
      if (va < 0) va += GSZ;
      s_vthr = va;
      s_useg2 = (i == 0) ? 1 : 0;
      s_ov1 = stopped ? act : -1;
      const int first = st_ai[0];
      s_ov2 = (!stopped && (nna == first)) ? first : -1;
      s_nol = stopped ? -1 : nna;
    }
    __syncthreads();
    // ---- gather input_q1 / input_q2 rows of h ----
    if (t < DD) {
      s_iq1[t] = h[((size_t)b * GSZ + s_iq1src) * DD + t];
    } else {
      const int d = t - DD;
      s_iq2[d] = h[((size_t)b * GSZ + s_iq2src) * DD + d];
    }
    __syncthreads();
  }

  if (t == 0) {
    if (!s_stopped) st_kar[KMAX - 1] = st_kal[KMAX];
    int* row = out_actions + b * (3 * KMAX);
    for (int j = 0; j < KMAX; ++j) {
      row[j] = st_ai[j];
      row[KMAX + j] = st_kal[j];
      row[2 * KMAX + j] = st_kar[j];
    }
    out_ll[b] = s_ll;
  }
}

// ---------------------------------------------------------------------------
extern "C" void kernel_launch(void* const* d_in, const int* in_sizes, int n_in,
                              void* d_out, int out_size, void* d_ws, size_t ws_size,
                              hipStream_t stream) {
  const float* h    = (const float*)d_in[0];
  const int* rec    = (const int*)d_in[1];
  /* d_in[2] = context2 (unused) */
  const int* vtim   = (const int*)d_in[3];
  const int* lastA  = (const int*)d_in[4];
  const int* fixA   = (const int*)d_in[5];
  const float* WK1  = (const float*)d_in[6];
  const float* WK2  = (const float*)d_in[7];
  const float* WK3  = (const float*)d_in[8];
  const float* WK4  = (const float*)d_in[9];
  const float* WQ1  = (const float*)d_in[10];
  const float* WQ2  = (const float*)d_in[11];
  const float* WQ3  = (const float*)d_in[12];
  const float* WQ4  = (const float*)d_in[13];
  const float* V1   = (const float*)d_in[14];
  const float* V2   = (const float*)d_in[15];
  const float* ih_w = (const float*)d_in[16];
  const float* ih_b = (const float*)d_in[17];
  const float* iniq = (const float*)d_in[18];
  const float* r1_wih = (const float*)d_in[19];
  const float* r1_whh = (const float*)d_in[20];
  const float* r1_bih = (const float*)d_in[21];
  const float* r1_bhh = (const float*)d_in[22];
  const float* r2_wih = (const float*)d_in[23];
  const float* r2_whh = (const float*)d_in[24];
  const float* r2_bih = (const float*)d_in[25];
  const float* r2_bhh = (const float*)d_in[26];

  const size_t KH = (size_t)BS * GSZ * DD;  // 16,384,000 elems
  __bf16* K1 = (__bf16*)d_ws;
  __bf16* K2 = K1 + KH;
  __bf16* K3 = K2 + KH;
  __bf16* K4 = K3 + KH;

  // Phase 1: WMMA bf16 key projections (M=128000 rows, 64 rows/block, 4 mats)
  dim3 g1((BS * GSZ) / 64, 4);
  kprep_wmma<<<g1, 128, 0, stream>>>(h, WK1, WK2, WK3, WK4, K1, K2, K3, K4);

  // Phase 2: sequential decode, one block per batch row
  int* out_actions = (int*)d_out;
  float* out_ll = (float*)d_out + BS * 3 * KMAX;
  kdecode<<<BS, 256, 0, stream>>>(h, rec, vtim, lastA, fixA,
                                  WQ1, WQ2, WQ3, WQ4, V1, V2,
                                  ih_w, ih_b, iniq,
                                  r1_wih, r1_whh, r1_bih, r1_bhh,
                                  r2_wih, r2_whh, r2_bih, r2_bhh,
                                  K1, K2, K3, K4, out_actions, out_ll);
}
